// FasterRCNN_48919677501805
// MI455X (gfx1250) — matrix operand order repaired and checked by reading, not compiled, not run on hardware
//
#include <hip/hip_runtime.h>
#include <hip/hip_bf16.h>
#include <math.h>

// ---------------------------------------------------------------------------
// Types for CDNA5 WMMA (wave32): v_wmma_f32_16x16x32_bf16
// ---------------------------------------------------------------------------
typedef __bf16 bf16_t;
typedef __attribute__((ext_vector_type(16))) __bf16 v16bf;
typedef __attribute__((ext_vector_type(8)))  __bf16 v8bf;
typedef __attribute__((ext_vector_type(8)))  float  v8f;

#define ROIS      128
#define P         7
#define CCH       512
#define HH        32
#define WW        32
#define DIN       (CCH * P * P)   // 25088
#define DH        4096
#define NHEAD     105             // 21 cls + 84 bbox
#define NHEAD_PAD 128
#define SLABS     8
#define LDSB_STRIDE 40            // 32 data + 8 pad bf16 per N-row (16B-aligned halves)

// ---------------------------------------------------------------------------
// 1) ROI adaptive max pool  ->  A1 bf16 [128][25088]
// ---------------------------------------------------------------------------
__global__ void __launch_bounds__(256)
roi_pool_kernel(const float* __restrict__ feats, const float* __restrict__ rois,
                bf16_t* __restrict__ A)
{
    int idx = blockIdx.x * 256 + threadIdx.x;      // over 128*512*49
    if (idx >= ROIS * CCH * P * P) return;
    int pw = idx % P;  int t = idx / P;
    int ph = t % P;    t /= P;
    int c  = t % CCH;  int n = t / CCH;

    const float* r = rois + n * 5;
    int x1 = (int)floorf(r[1] * 0.0625f);
    int y1 = (int)floorf(r[2] * 0.0625f);
    int x2 = (int)floorf(r[3] * 0.0625f);
    int y2 = (int)floorf(r[4] * 0.0625f);
    int h = y2 - y1 + 1, w = x2 - x1 + 1;
    int rs = y1 + (ph * h) / P;
    int re = y1 + ((ph + 1) * h + (P - 1)) / P;
    int cs = x1 + (pw * w) / P;
    int ce = x1 + ((pw + 1) * w + (P - 1)) / P;

    const float* f = feats + c * (HH * WW);
    float m = -INFINITY;
    for (int rr = rs; rr < re; ++rr) {
        int rc = min(max(rr, 0), HH - 1);
        for (int cc = cs; cc < ce; ++cc) {
            int cl = min(max(cc, 0), WW - 1);
            m = fmaxf(m, f[rc * WW + cl]);
        }
    }
    A[(size_t)n * DIN + c * (P * P) + ph * P + pw] = (bf16_t)m;
}

// ---------------------------------------------------------------------------
// 2) Split-K bf16 WMMA GEMM: partial[slab] = A[128 x Kslab] * B[Kslab x N]
//    A: bf16 row-major (lda = K). B: f32 row-major [K][LDB] (LDB compile-time,
//    so weight-streaming uses immediate-offset clauses), converted f32->bf16
//    while staging N-major into LDS. Weights are loaded non-temporally (they
//    are dead after one use; keeps activations + partials hot in L2).
//    Block = 256 thr (8 waves); each block computes a 128x128 C tile.
// ---------------------------------------------------------------------------
template<int LDB>
__global__ void __launch_bounds__(256)
gemm_bf16_splitk(const bf16_t* __restrict__ A, int lda,
                 const float* __restrict__ B,
                 float* __restrict__ partials, int k_per_slab)
{
    __shared__ bf16_t ldsB[128 * LDSB_STRIDE];

    const int n0    = blockIdx.x * 128;
    const int slab  = blockIdx.y;
    const int kbeg  = slab * k_per_slab;
    const int steps = k_per_slab >> 5;

    const int t    = threadIdx.x;
    const int wave = t >> 5;
    const int lane = t & 31;
    const int m0   = wave * 16;

    // A fragment addressing (16-bit A 16x32 layout):
    //   lanes 0-15 : K 0..7 (chunk0), 16..23 (chunk1); lanes 16-31: K 8..15, 24..31
    const int rowA = m0 + (lane & 15);
    const int selA = (lane >> 4) * 8;
    const bf16_t* Arow = A + (size_t)rowA * lda + kbeg + selA;

    // B fragment addressing (from N-major LDS tile):
    //   lanes 0-15: N=lane, K 0..7, 16..23 ; lanes 16-31: N=lane-16, K 8..15, 24..31
    const bf16_t* Bfrag = ldsB + (lane & 15) * LDSB_STRIDE + (lane >> 4) * 8;

    // B staging: thread owns column n = t&127, contiguous K-run of 16.
    // Wave-coalesced along n; 16 immediate-offset loads; 2x ds_store_b128.
    const int bn    = t & 127;
    const int khalf = (t >> 7) * 16;
    const float* Bp = B + (size_t)(kbeg + khalf) * LDB + n0 + bn;
    bf16_t* ldsW = ldsB + bn * LDSB_STRIDE + khalf;

    v8f acc[8];
    #pragma unroll
    for (int i = 0; i < 8; ++i) acc[i] = (v8f){0.f,0.f,0.f,0.f,0.f,0.f,0.f,0.f};

    for (int s = 0; s < steps; ++s) {
        // Issue all 16 weight loads up front (distinct regs -> one clause,
        // latency overlapped with previous step's LDS reads until barrier).
        float v[16];
        #pragma unroll
        for (int i = 0; i < 16; ++i)
            v[i] = __builtin_nontemporal_load(Bp + (size_t)i * LDB);
        if (s + 1 < steps)
            __builtin_prefetch(Bp + (size_t)32 * LDB, 0, 1);
        Bp += (size_t)32 * LDB;

        __syncthreads();   // all waves done reading previous LDS tile

        v8bf lo, hi;
        #pragma unroll
        for (int j = 0; j < 8; ++j) {
            lo[j] = (bf16_t)v[j];
            hi[j] = (bf16_t)v[8 + j];
        }
        *(v8bf*)(ldsW)     = lo;   // ds_store_b128
        *(v8bf*)(ldsW + 8) = hi;   // ds_store_b128

        __syncthreads();

        // A fragment: two contiguous 16B global loads (L2-resident)
        v8bf alo = *(const v8bf*)(Arow + s * 32);
        v8bf ahi = *(const v8bf*)(Arow + s * 32 + 16);
        v16bf a = __builtin_shufflevector(alo, ahi,
                    0,1,2,3,4,5,6,7,8,9,10,11,12,13,14,15);

        #pragma unroll
        for (int tt = 0; tt < 8; ++tt) {
            const bf16_t* bp = Bfrag + tt * 16 * LDSB_STRIDE;
            v8bf blo = *(const v8bf*)(bp);
            v8bf bhi = *(const v8bf*)(bp + 16);
            v16bf b = __builtin_shufflevector(blo, bhi,
                        0,1,2,3,4,5,6,7,8,9,10,11,12,13,14,15);
            acc[tt] = __builtin_amdgcn_wmma_f32_16x16x32_bf16(
                          false, a, false, b, (short)0, acc[tt], false, false);
        }
    }

    // C layout: VGPR g -> M = m0 + (lane<16 ? g : 8+g), N = n0 + tt*16 + (lane&15)
    float* Pp = partials + (size_t)slab * ROIS * LDB;
    const int moff = m0 + (lane >> 4) * 8;
    #pragma unroll
    for (int tt = 0; tt < 8; ++tt) {
        int n = n0 + tt * 16 + (lane & 15);
        #pragma unroll
        for (int g = 0; g < 8; ++g) {
            Pp[(size_t)(moff + g) * LDB + n] = acc[tt][g];
        }
    }
}

// ---------------------------------------------------------------------------
// 3) Deterministic split-K reduce + bias + ReLU -> bf16 activations
// ---------------------------------------------------------------------------
__global__ void __launch_bounds__(256)
reduce_bias_relu(const float* __restrict__ partials, int N,
                 const float* __restrict__ bias, bf16_t* __restrict__ out)
{
    int idx = blockIdx.x * 256 + threadIdx.x;    // over 128*N
    if (idx >= ROIS * N) return;
    int n = idx % N;
    float v = bias[n];
    #pragma unroll
    for (int s = 0; s < SLABS; ++s) v += partials[(size_t)s * ROIS * N + idx];
    out[idx] = (bf16_t)fmaxf(v, 0.f);
}

// ---------------------------------------------------------------------------
// 4) Pack head weights: WH f32 [4096][128] = [Ws | Wb | zero-pad]
// ---------------------------------------------------------------------------
__global__ void __launch_bounds__(256)
pack_head_weights(const float* __restrict__ Ws, const float* __restrict__ Wb,
                  float* __restrict__ WH)
{
    int idx = blockIdx.x * 256 + threadIdx.x;    // over 4096*128
    if (idx >= DH * NHEAD_PAD) return;
    int j = idx % NHEAD_PAD;
    int k = idx / NHEAD_PAD;
    float v = 0.f;
    if (j < 21)         v = Ws[k * 21 + j];
    else if (j < NHEAD) v = Wb[k * 84 + (j - 21)];
    WH[idx] = v;
}

// ---------------------------------------------------------------------------
// 5) Head finalize: reduce split-K, +bias, softmax(21) || bbox(84) -> out
// ---------------------------------------------------------------------------
__global__ void __launch_bounds__(128)
head_finalize(const float* __restrict__ partials,
              const float* __restrict__ bs, const float* __restrict__ bb,
              float* __restrict__ out)
{
    int m = blockIdx.x;          // ROI row
    int j = threadIdx.x;         // 0..127
    __shared__ float vals[NHEAD];
    __shared__ float red[2];

    if (j < NHEAD) {
        float v = (j < 21) ? bs[j] : bb[j - 21];
        #pragma unroll
        for (int s = 0; s < SLABS; ++s)
            v += partials[(size_t)s * ROIS * NHEAD_PAD + m * NHEAD_PAD + j];
        vals[j] = v;
    }
    __syncthreads();
    if (j == 0) {
        float mx = vals[0];
        for (int i = 1; i < 21; ++i) mx = fmaxf(mx, vals[i]);
        float sum = 0.f;
        for (int i = 0; i < 21; ++i) sum += expf(vals[i] - mx);
        red[0] = mx; red[1] = sum;
    }
    __syncthreads();
    if (j < NHEAD) {
        float v = vals[j];
        out[m * NHEAD + j] = (j < 21) ? expf(v - red[0]) / red[1] : v;
    }
}

// ---------------------------------------------------------------------------
// Launch
// ---------------------------------------------------------------------------
extern "C" void kernel_launch(void* const* d_in, const int* in_sizes, int n_in,
                              void* d_out, int out_size, void* d_ws, size_t ws_size,
                              hipStream_t stream)
{
    const float* features = (const float*)d_in[0];
    const float* rois     = (const float*)d_in[1];
    const float* W6       = (const float*)d_in[2];
    const float* b6       = (const float*)d_in[3];
    const float* W7       = (const float*)d_in[4];
    const float* b7       = (const float*)d_in[5];
    const float* Ws       = (const float*)d_in[6];
    const float* bs       = (const float*)d_in[7];
    const float* Wb       = (const float*)d_in[8];
    const float* bb       = (const float*)d_in[9];
    float* out = (float*)d_out;

    char* ws = (char*)d_ws;
    bf16_t* A1    = (bf16_t*)(ws);                       // 128*25088*2 = 6,422,528
    bf16_t* x1    = (bf16_t*)(ws + 6422528);             // 128*4096*2  = 1,048,576
    bf16_t* x2    = (bf16_t*)(ws + 7471104);             // 128*4096*2  = 1,048,576
    float*  WH    = (float*) (ws + 8519680);             // 4096*128*4  = 2,097,152
    float*  parts = (float*) (ws + 10616832);            // 8*128*4096*4 = 16,777,216

    // 1) ROI max pool -> A1 (bf16)
    roi_pool_kernel<<<(ROIS * CCH * P * P + 255) / 256, 256, 0, stream>>>(
        features, rois, A1);

    // (independent) pack head weights
    pack_head_weights<<<(DH * NHEAD_PAD + 255) / 256, 256, 0, stream>>>(Ws, Wb, WH);

    // 2) FC6: 128x25088 @ 25088x4096, split-K over 8 slabs of 3136
    gemm_bf16_splitk<DH><<<dim3(DH / 128, SLABS), 256, 0, stream>>>(
        A1, DIN, W6, parts, DIN / SLABS);
    reduce_bias_relu<<<(ROIS * DH + 255) / 256, 256, 0, stream>>>(parts, DH, b6, x1);

    // 3) FC7: 128x4096 @ 4096x4096, split-K over 8 slabs of 512
    gemm_bf16_splitk<DH><<<dim3(DH / 128, SLABS), 256, 0, stream>>>(
        x1, DH, W7, parts, DH / SLABS);
    reduce_bias_relu<<<(ROIS * DH + 255) / 256, 256, 0, stream>>>(parts, DH, b7, x2);

    // 4) Heads: 128x4096 @ 4096x128(packed), split-K over 8 slabs of 512
    gemm_bf16_splitk<NHEAD_PAD><<<dim3(1, SLABS), 256, 0, stream>>>(
        x2, DH, WH, parts, DH / SLABS);

    // 5) softmax + concat -> out [128][105]
    head_finalize<<<ROIS, 128, 0, stream>>>(parts, bs, bb, out);
}
